// EmotionGateformerLoss_618475291417
// MI455X (gfx1250) — matrix operand order repaired and checked by reading, not compiled, not run on hardware
//
#include <hip/hip_runtime.h>

// ---- problem constants (from reference setup_inputs) ----
#define B_   32
#define S_   1024
#define D_   3
#define DS_  256
#define ST_  (S_ / 16)   // 64 s-tiles of 16 rows

typedef __attribute__((ext_vector_type(16))) _Float16 v16h;
typedef __attribute__((ext_vector_type(8)))  _Float16 v8h;
typedef __attribute__((ext_vector_type(8)))  float    v8f;

// acc layout (floats): [0]=n [1]=huber_sum [2..4]=Sp [5..7]=Sg
// [8..10]=Spp [11..13]=Sgg [14..16]=Spg [17]=delta_sum [18]=delta_cnt

// -------------------------------------------------------------------------
// Kernel 1: L2-normalize each 256-dim speaker row, convert to f16.
// One wave (32 lanes) per row, 8 strided elements per lane.
// -------------------------------------------------------------------------
__global__ void egl_normalize(const float* __restrict__ spk,
                              _Float16* __restrict__ out) {
  const int row  = (blockIdx.x * blockDim.x + threadIdx.x) >> 5;
  const int lane = threadIdx.x & 31;
  if (row >= B_ * S_) return;
  const float* rp = spk + (size_t)row * DS_;
  float v[8];
  float ss = 0.f;
#pragma unroll
  for (int i = 0; i < 8; ++i) {
    v[i] = rp[lane + 32 * i];
    ss += v[i] * v[i];
  }
#pragma unroll
  for (int off = 16; off > 0; off >>= 1) ss += __shfl_xor(ss, off, 32);
  const float scale = 1.0f / fmaxf(sqrtf(ss), 1e-12f);
  _Float16* op = out + (size_t)row * DS_;
#pragma unroll
  for (int i = 0; i < 8; ++i) op[lane + 32 * i] = (_Float16)(v[i] * scale);
}

// -------------------------------------------------------------------------
// Kernel 2: masked Huber sum + masked raw moments for CCC.
// -------------------------------------------------------------------------
__global__ void egl_moments(const float* __restrict__ pred,
                            const float* __restrict__ gt,
                            const unsigned char* __restrict__ pad,
                            float* __restrict__ acc) {
  const int i = blockIdx.x * blockDim.x + threadIdx.x;
  float loc[17];
#pragma unroll
  for (int k = 0; k < 17; ++k) loc[k] = 0.f;
  if (i < B_ * S_) {
    if (!pad[i]) {  // active
      float hub = 0.f;
#pragma unroll
      for (int d = 0; d < D_; ++d) {
        const float p = pred[i * D_ + d];
        const float g = gt[i * D_ + d];
        const float e = p - g;
        const float ae = fabsf(e);
        hub += (ae < 1.f) ? 0.5f * e * e : (ae - 0.5f);
        loc[2 + d]  = p;
        loc[5 + d]  = g;
        loc[8 + d]  = p * p;
        loc[11 + d] = g * g;
        loc[14 + d] = p * g;
      }
      loc[0] = 1.f;
      loc[1] = hub;
    }
  }
  __shared__ float sm[17][8];
  const int lane = threadIdx.x & 31;
  const int wid  = threadIdx.x >> 5;
#pragma unroll
  for (int k = 0; k < 17; ++k) {
    float v = loc[k];
#pragma unroll
    for (int off = 16; off > 0; off >>= 1) v += __shfl_xor(v, off, 32);
    if (lane == 0) sm[k][wid] = v;
  }
  __syncthreads();
  if (threadIdx.x == 0) {
#pragma unroll
    for (int k = 0; k < 17; ++k) {
      float s = 0.f;
#pragma unroll
      for (int w = 0; w < 8; ++w) s += sm[k][w];
      atomicAdd(&acc[k], s);
    }
  }
}

// -------------------------------------------------------------------------
// Kernel 3: WMMA similarity scan. One wave per (b, 16-row s-tile).
// A fragments (16x32 f16 per K-chunk) loaded once; B fragments per t-tile.
// Running best-t per row (t ascending => last write wins == argmax).
// -------------------------------------------------------------------------
__global__ __launch_bounds__(32) void egl_prev(const _Float16* __restrict__ spk,
                                               const unsigned char* __restrict__ pad,
                                               int* __restrict__ prev) {
  const int b  = blockIdx.x >> 6;    // /ST_
  const int st = blockIdx.x & 63;
  const int s0 = st * 16;
  const int l  = threadIdx.x;
  const int m  = l & 15;
  const int kh = l >> 4;

  const _Float16* base = spk + (size_t)b * S_ * DS_;
  const unsigned char* pb = pad + b * S_;

  // ---- A fragments: A[m][k] = spk[s0+m][k]; ISA 16-bit A 16x32 layout:
  // lane half kh: a[0..7] <- K = 32c + 8*kh .. +7 ; a[8..15] <- K = 32c+16+8*kh .. +7
  v16h Afrag[8];
  {
    const _Float16* rp = base + (size_t)(s0 + m) * DS_;
#pragma unroll
    for (int c = 0; c < 8; ++c) {
      const v8h h0 = *(const v8h*)(rp + 32 * c + 8 * kh);
      const v8h h1 = *(const v8h*)(rp + 32 * c + 16 + 8 * kh);
#pragma unroll
      for (int e = 0; e < 8; ++e) {
        Afrag[c][e]     = h0[e];
        Afrag[c][8 + e] = h1[e];
      }
    }
  }

  // active bits for this lane's 8 output rows (C layout: M = r + 8*kh)
  unsigned rowact = 0;
#pragma unroll
  for (int r = 0; r < 8; ++r)
    if (!pb[s0 + r + 8 * kh]) rowact |= (1u << r);

  int best[8];
#pragma unroll
  for (int r = 0; r < 8; ++r) best[r] = -1;

  for (int tt = 0; tt <= st; ++tt) {
    const int t0   = tt * 16;
    const int tcol = t0 + m;                 // this lane's output column
    const bool colact = (pb[tcol] == 0);
    const _Float16* cp = base + (size_t)tcol * DS_;

    v8f accv = {};
#pragma unroll
    for (int c = 0; c < 8; ++c) {
      // B layout (32x16 f16): lane n=m holds K = 32c + 8*kh + e (e<8), 32c + 16 + 8*kh + (e-8) (e>=8)
      const v8h h0 = *(const v8h*)(cp + 32 * c + 8 * kh);
      const v8h h1 = *(const v8h*)(cp + 32 * c + 16 + 8 * kh);
      v16h Bfrag;
#pragma unroll
      for (int e = 0; e < 8; ++e) {
        Bfrag[e]     = h0[e];
        Bfrag[8 + e] = h1[e];
      }
      accv = __builtin_amdgcn_wmma_f32_16x16x32_f16(
          false, Afrag[c], false, Bfrag, (short)0, accv, false, false);
    }

    if (colact) {
#pragma unroll
      for (int r = 0; r < 8; ++r) {
        const int s = s0 + r + 8 * kh;
        if (((rowact >> r) & 1u) && (tcol < s) && (accv[r] > 0.99f))
          best[r] = tcol;  // t ascending: later tiles overwrite => max t
      }
    }
  }

  // max-reduce each row across its 16-lane half group (xor<16 stays in group)
#pragma unroll
  for (int r = 0; r < 8; ++r) {
    int v = best[r];
#pragma unroll
    for (int off = 1; off < 16; off <<= 1) {
      const int o = __shfl_xor(v, off, 32);
      v = (o > v) ? o : v;
    }
    if (m == 0) prev[b * S_ + s0 + r + 8 * kh] = v;
  }
}

// -------------------------------------------------------------------------
// Kernel 4: delta term gather + reduce.
// -------------------------------------------------------------------------
__global__ void egl_delta(const float* __restrict__ pred,
                          const float* __restrict__ gt,
                          const int* __restrict__ prev,
                          float* __restrict__ acc) {
  const int i = blockIdx.x * blockDim.x + threadIdx.x;
  float ds = 0.f, cnt = 0.f;
  if (i < B_ * S_) {
    const int p = prev[i];
    if (p >= 0) {
      const int b = i / S_;
      const int j = b * S_ + p;
      float sum = 0.f;
#pragma unroll
      for (int d = 0; d < D_; ++d) {
        const float diff = (pred[i * D_ + d] - pred[j * D_ + d]) -
                           (gt[i * D_ + d] - gt[j * D_ + d]);
        sum += diff * diff;
      }
      ds = sum;
      cnt = 1.f;
    }
  }
#pragma unroll
  for (int off = 16; off > 0; off >>= 1) {
    ds += __shfl_xor(ds, off, 32);
    cnt += __shfl_xor(cnt, off, 32);
  }
  if ((threadIdx.x & 31) == 0) {
    atomicAdd(&acc[17], ds);
    atomicAdd(&acc[18], cnt);
  }
}

// -------------------------------------------------------------------------
// Kernel 5: finalize scalar loss.
// -------------------------------------------------------------------------
__global__ void egl_final(const float* __restrict__ acc, float* __restrict__ out) {
  if (threadIdx.x == 0 && blockIdx.x == 0) {
    const float n   = acc[0];
    const float l_huber = acc[1] / (n * (float)D_ + 1e-8f);
    float ccc_sum = 0.f;
#pragma unroll
    for (int d = 0; d < D_; ++d) {
      const float Sp = acc[2 + d], Sg = acc[5 + d];
      const float Spp = acc[8 + d], Sgg = acc[11 + d], Spg = acc[14 + d];
      const float mu_p = Sp / n, mu_g = Sg / n;
      const float var_p = (Spp - n * mu_p * mu_p) / (n - 1.0f);
      const float var_g = (Sgg - n * mu_g * mu_g) / (n - 1.0f);
      const float cov   = (Spg - n * mu_p * mu_g) / n;
      const float dmu   = mu_p - mu_g;
      ccc_sum += 2.0f * cov / (var_p + var_g + dmu * dmu + 1e-8f);
    }
    const float l_ccc   = 1.0f - ccc_sum / (float)D_;
    const float l_delta = acc[17] / (acc[18] + 1e-8f) / (float)D_;
    out[0] = 1.0f * l_huber + 1.0f * l_ccc + 5.0f * l_delta;
  }
}

// -------------------------------------------------------------------------
extern "C" void kernel_launch(void* const* d_in, const int* in_sizes, int n_in,
                              void* d_out, int out_size, void* d_ws, size_t ws_size,
                              hipStream_t stream) {
  (void)in_sizes; (void)n_in; (void)out_size; (void)ws_size;
  const float* pred = (const float*)d_in[0];
  const float* gt   = (const float*)d_in[1];
  const float* dspk = (const float*)d_in[2];
  const unsigned char* pad = (const unsigned char*)d_in[3];
  float* out = (float*)d_out;

  char* ws = (char*)d_ws;
  float* acc     = (float*)ws;                               // 19 floats (pad 128B)
  int* prev      = (int*)(ws + 128);                         // B*S ints (128 KB)
  _Float16* spkh = (_Float16*)(ws + 128 + B_ * S_ * sizeof(int));  // 16 MB f16

  hipMemsetAsync(acc, 0, 128, stream);

  // one wave per row, 8 waves per block
  egl_normalize<<<(B_ * S_) / 8, 256, 0, stream>>>(dspk, spkh);
  egl_moments<<<(B_ * S_ + 255) / 256, 256, 0, stream>>>(pred, gt, pad, acc);
  egl_prev<<<B_ * ST_, 32, 0, stream>>>(spkh, pad, prev);
  egl_delta<<<(B_ * S_ + 255) / 256, 256, 0, stream>>>(pred, gt, prev, acc);
  egl_final<<<1, 1, 0, stream>>>(acc, out);
}
